// get_model_82970178224395
// MI455X (gfx1250) — compile-verified, matrix operand order checked by reading, not run on hardware
//
#include <hip/hip_runtime.h>
#include <math.h>

#define BB   8
#define NP   2048
#define DD   256
#define CC   128
#define NCLSv 50
#define NT   (BB*NP)

typedef __attribute__((ext_vector_type(16))) __bf16 v16bf;
typedef __attribute__((ext_vector_type(8)))  __bf16 v8bf;
typedef __attribute__((ext_vector_type(8)))  float  v8f;

// ---------------------------------------------------------------------------
// Fused epilogue descriptor:
//   r = sign*acc (+ prevPre) ; r = r*scale[ch] + bias[ch] ; relu ; (+ prevPost)
//   stores f32 (C) and/or bf16 (Cbf) copies with independent batch strides.
struct Epi {
    float sign;
    const float* prevPre;  long ppB;
    const float* scale;
    const float* bias;
    int  chRow;            // 1: channel index = row m, 0: = col n
    int  relu;
    const float* prevPost; long poB;
    float*  C;   long cB;
    __bf16* Cbf; long cbfB;
};

__device__ __forceinline__ void store_tile(const v8f& acc, int b, int m0, int cn,
                                           int lane, int M, int Nn, const Epi& e)
{
#pragma unroll
    for (int r = 0; r < 8; ++r) {
        int cm = m0 + r + 8 * (lane >> 4);
        if (cm < M) {
            long idx = (long)cm * Nn + cn;
            float v = e.sign * acc[r];
            if (e.prevPre)  v += e.prevPre[(long)b * e.ppB + idx];
            int ch = e.chRow ? cm : cn;
            if (e.scale) v *= e.scale[ch];
            if (e.bias)  v += e.bias[ch];
            if (e.relu)  v = fmaxf(v, 0.0f);
            if (e.prevPost) v += e.prevPost[(long)b * e.poB + idx];
            if (e.C)   e.C  [(long)b * e.cB   + idx] = v;
            if (e.Cbf) e.Cbf[(long)b * e.cbfB + idx] = (__bf16)v;
        }
    }
}

// async copy of 16 bytes: global(p + byteOff) -> LDS(ldsOff + byteOff).
// completion is tracked by ASYNCcnt.
#define ASYNC_B128(ldsOff, p, byteOff)                                          \
    asm volatile("global_load_async_to_lds_b128 %0, %1, off offset:" #byteOff   \
                 :: "v"(ldsOff), "v"(p) : "memory")

// ---------------------------------------------------------------------------
// Batched strided bf16 WMMA GEMM: C[b](M x Nn) = A[b] * B[b]
// A(m,k) = Aq[b*sAb + m*sAm + k*sAk] ; B(k,n) = Bq[b*sBb + k*sBk + n*sBn]
// Call-site guarantees: K % 32 == 0, Nn % 32 == 0; M ragged only for classifier.
// MA/MB: 0 = operand k-contiguous (stride_k==1), direct b128 fragment loads
//        1 = operand m/n-contiguous: async global->LDS (ASYNCcnt) staging
// Each wave computes a 16(m) x 32(n) tile: one A fragment feeds two WMMAs.
template<int MA, int MB>
__global__ __launch_bounds__(128)
void wmma_gemm(const __bf16* __restrict__ Aq, const __bf16* __restrict__ Bq,
               int batch, int M, int Nn, int K,
               long sAb, long sAm, long sAk,
               long sBb, long sBk, long sBn,
               int tilesM, int tilesN, Epi e)
{
    constexpr int PW = ((MA == 1) ? 512 : 0) + ((MB == 1) ? 1024 : 0);
    __shared__ __bf16 ldsbuf[PW ? 4 * PW : 4];
    const int wave = threadIdx.x >> 5;
    const int lane = threadIdx.x & 31;
    __bf16* la = ldsbuf + wave * PW;
    __bf16* lb = la + ((MA == 1) ? 512 : 0);

    const long tilesPerB = (long)tilesM * tilesN;
    const long gtile = (long)blockIdx.x * 4 + wave;
    if (gtile >= tilesPerB * batch) return;          // wave-uniform
    const int  b  = (int)(gtile / tilesPerB);
    const long t  = gtile - (long)b * tilesPerB;
    const int  tm = (int)(t / tilesN);
    const int  tn = (int)(t - (long)tm * tilesN);
    const int  m0 = tm * 16, n0 = tn * 32;

    const __bf16* Ab = Aq + (long)b * sAb;
    const __bf16* Bb = Bq + (long)b * sBb;
    const int l15 = lane & 15;
    const int kh  = (lane >> 4) * 8;
    int am = m0 + l15; if (am >= M) am = M - 1;      // clamp: OOB rows never stored

    // per-lane marching pointers (no per-iteration index arithmetic)
    const __bf16* pA; long dA;
    if constexpr (MA == 0) { pA = Ab + (long)am * sAm + kh; dA = 32; }
    else                   { pA = Ab + m0 + (long)lane * sAk; dA = 32 * sAk; }
    const __bf16* pB; long dB;
    if constexpr (MB == 0) { pB = Bb + (long)(n0 + l15) * sBn + kh; dB = 32; }
    else                   { pB = Bb + (long)lane * sBk + n0; dB = 32 * sBk; }
    const long nb16 = (MB == 0) ? 16 * sBn : 0;      // second-column offset (direct)

    const unsigned laOff = (unsigned)(size_t)&la[lane * 16];   // LDS byte addr (addr[31:0])
    const unsigned lbOff = (unsigned)(size_t)&lb[lane * 32];

    v8f acc0 = {}, acc1 = {};
    for (int it = K >> 5; it > 0; --it) {
        v16bf af, b0, b1;
        // ---- A fragment: lane holds row m = l15, K = kh..kh+7 and kh+16..kh+23
        if constexpr (MA == 0) {
            __builtin_prefetch(pA + dA, 0, 1);
            v8bf lo = *(const v8bf*)pA;
            v8bf hi = *(const v8bf*)(pA + 16);
#pragma unroll
            for (int j = 0; j < 8; ++j) { af[j] = lo[j]; af[8 + j] = hi[j]; }
        } else {
            // A m-contiguous: async-copy one k-column (16 rows, 32B) into LDS [k][m]
            ASYNC_B128(laOff, pA, 0);
            ASYNC_B128(laOff, pA, 16);
            asm volatile("s_wait_asynccnt 0x0" ::: "memory");
#pragma unroll
            for (int j = 0; j < 8; ++j) {
                af[j]     = la[(kh + j) * 16 + l15];
                af[8 + j] = la[(kh + 16 + j) * 16 + l15];
            }
        }
        // ---- B fragments (two 16-wide column tiles)
        if constexpr (MB == 0) {
            __builtin_prefetch(pB + dB, 0, 1);
            v8bf lo0 = *(const v8bf*)pB;
            v8bf hi0 = *(const v8bf*)(pB + 16);
            v8bf lo1 = *(const v8bf*)(pB + nb16);
            v8bf hi1 = *(const v8bf*)(pB + nb16 + 16);
#pragma unroll
            for (int j = 0; j < 8; ++j) {
                b0[j] = lo0[j]; b0[8 + j] = hi0[j];
                b1[j] = lo1[j]; b1[8 + j] = hi1[j];
            }
        } else {
            // B n-contiguous: async-copy one k-row (32 cols, 64B) into LDS [k][n]
            ASYNC_B128(lbOff, pB, 0);
            ASYNC_B128(lbOff, pB, 16);
            ASYNC_B128(lbOff, pB, 32);
            ASYNC_B128(lbOff, pB, 48);
            asm volatile("s_wait_asynccnt 0x0" ::: "memory");
#pragma unroll
            for (int j = 0; j < 8; ++j) {
                b0[j]     = lb[(kh + j) * 32 + l15];
                b0[8 + j] = lb[(kh + 16 + j) * 32 + l15];
                b1[j]     = lb[(kh + j) * 32 + 16 + l15];
                b1[8 + j] = lb[(kh + 16 + j) * 32 + 16 + l15];
            }
        }
        acc0 = __builtin_amdgcn_wmma_f32_16x16x32_bf16(false, af, false, b0,
                                                       (short)0, acc0, false, false);
        acc1 = __builtin_amdgcn_wmma_f32_16x16x32_bf16(false, af, false, b1,
                                                       (short)0, acc1, false, false);
        pA += dA; pB += dB;
    }
    store_tile(acc0, b, m0, n0 + l15,      lane, M, Nn, e);
    store_tile(acc1, b, m0, n0 + 16 + l15, lane, M, Nn, e);
}

// ---------------------------------------------------------------------------
// VecKM phase: eA = exp(i * pts@A), eB = exp(i * pts@Bmat), stored bf16 (B,N,D)
__global__ void k_phase(const float* __restrict__ xyz, const float* __restrict__ A,
                        const float* __restrict__ Bm,
                        __bf16* eAr, __bf16* eAi, __bf16* eBr, __bf16* eBi)
{
    long i = (long)blockIdx.x * blockDim.x + threadIdx.x;
    if (i >= (long)NT * DD) return;
    int d = (int)(i % DD);
    long bn = i / DD;
    int n = (int)(bn % NP);
    int b = (int)(bn / NP);
    const float* X = xyz + (long)b * 3 * NP;
    float x = X[n], y = X[NP + n], z = X[2 * NP + n];
    float pa = x * A[d]  + y * A[DD + d]  + z * A[2 * DD + d];
    float pb = x * Bm[d] + y * Bm[DD + d] + z * Bm[2 * DD + d];
    float s, c;
    __sincosf(pa, &s, &c); eAr[i] = (__bf16)c; eAi[i] = (__bf16)s;
    __sincosf(pb, &s, &c); eBr[i] = (__bf16)c; eBi[i] = (__bf16)s;
}

__global__ void k_f2bf(const float* __restrict__ in, __bf16* __restrict__ out, long n)
{
    long i = (long)blockIdx.x * blockDim.x + threadIdx.x;
    if (i < n) out[i] = (__bf16)in[i];
}

__global__ void k_neg(const float* __restrict__ in, float* __restrict__ out, int n)
{
    int i = blockIdx.x * blockDim.x + threadIdx.x;
    if (i < n) out[i] = -in[i];
}

// scale = g/sqrt(1+eps); fb = (cb ? cb*scale : 0) + b   (BN running stats 0/1)
__global__ void k_prep_sb(const float* g, const float* b, const float* cb,
                          float* scale, float* fb, int n)
{
    int i = blockIdx.x * blockDim.x + threadIdx.x;
    if (i >= n) return;
    float s = g[i] * rsqrtf(1.0f + 1e-5f);
    scale[i] = s;
    fb[i] = (cb ? cb[i] * s : 0.0f) + b[i];
}

__global__ void k_reweight(const float* __restrict__ cls, const float* __restrict__ lbl,
                           float* __restrict__ rw)
{
    int i = blockIdx.x * blockDim.x + threadIdx.x;
    if (i >= BB * DD) return;
    int d = i % DD, b = i / DD;
    float s = 0.0f;
    for (int k = 0; k < 16; ++k) s += cls[b * 16 + k] * lbl[k * DD + d];
    rw[i] = s;
}

// normalize complex G rows to |G| = sqrt(D) = 16, emit bf16
__global__ void k_vnorm(const float* __restrict__ Gr, const float* __restrict__ Gi,
                        __bf16* __restrict__ oR, __bf16* __restrict__ oI)
{
    __shared__ float red[256];
    long t = blockIdx.x;
    int d = threadIdx.x;
    float r = Gr[t * DD + d], im = Gi[t * DD + d];
    red[d] = r * r + im * im;
    __syncthreads();
    for (int s = 128; s > 0; s >>= 1) { if (d < s) red[d] += red[d + s]; __syncthreads(); }
    float sc = 16.0f * rsqrtf(red[0] + 1e-30f);
    oR[t * DD + d] = (__bf16)(r * sc);
    oI[t * DD + d] = (__bf16)(im * sc);
}

// x(B,D,N) = (re^2+im^2) * reweight[b,d]  (token-major F -> channel-major bf16)
__global__ void k_magrw(const float* __restrict__ Fr, const float* __restrict__ Fi,
                        const float* __restrict__ rw, __bf16* __restrict__ out)
{
    long i = (long)blockIdx.x * blockDim.x + threadIdx.x;
    if (i >= (long)BB * DD * NP) return;
    int n = (int)(i % NP);
    long bd = i / NP;
    int d = (int)(bd % DD);
    int b = (int)(bd / DD);
    long ti = ((long)b * NP + n) * DD + d;
    float re = Fr[ti], im = Fi[ti];
    out[i] = (__bf16)((re * re + im * im) * rw[b * DD + d]);
}

__global__ void k_softmax_rows(float* __restrict__ e)
{
    __shared__ float red[256];
    float* row = e + (long)blockIdx.x * NP;
    int t = threadIdx.x;
    float mx = -1e30f;
    for (int i = t; i < NP; i += 256) mx = fmaxf(mx, row[i]);
    red[t] = mx; __syncthreads();
    for (int s = 128; s > 0; s >>= 1) { if (t < s) red[t] = fmaxf(red[t], red[t + s]); __syncthreads(); }
    mx = red[0]; __syncthreads();
    float sum = 0.0f;
    for (int i = t; i < NP; i += 256) { float v = __expf(row[i] - mx); row[i] = v; sum += v; }
    red[t] = sum; __syncthreads();
    for (int s = 128; s > 0; s >>= 1) { if (t < s) red[t] += red[t + s]; __syncthreads(); }
    float inv = 1.0f / red[0];
    for (int i = t; i < NP; i += 256) row[i] *= inv;
}

__global__ void k_colsum(const float* __restrict__ attn, float* __restrict__ cs)
{
    int b = blockIdx.y;
    int m = blockIdx.x * 256 + threadIdx.x;
    const float* p = attn + (long)b * NP * NP + m;
    float s = 0.0f;
    for (int n = 0; n < NP; ++n) s += p[(long)n * NP];
    cs[b * NP + m] = s;
}

__global__ void k_renorm(const float* __restrict__ attn, const float* __restrict__ cs,
                         __bf16* __restrict__ out)
{
    long i = (long)blockIdx.x * blockDim.x + threadIdx.x;
    if (i >= (long)BB * NP * NP) return;
    int m = (int)(i % NP);
    int b = (int)(i / ((long)NP * NP));
    out[i] = (__bf16)(attn[i] / (1e-9f + cs[b * NP + m]));
}

__global__ void k_logsm(const float* __restrict__ c2o, float* __restrict__ out)
{
    int i = blockIdx.x * blockDim.x + threadIdx.x;
    if (i >= NT) return;
    int b = i / NP, n = i % NP;
    const float* p = c2o + (long)b * NCLSv * NP + n;
    float mx = -1e30f;
    for (int o = 0; o < NCLSv; ++o) mx = fmaxf(mx, p[(long)o * NP]);
    float s = 0.0f;
    for (int o = 0; o < NCLSv; ++o) s += __expf(p[(long)o * NP] - mx);
    float l = mx + __logf(s);
    float* q = out + (long)i * NCLSv;
    for (int o = 0; o < NCLSv; ++o) q[o] = p[(long)o * NP] - l;
}

// ---------------------------------------------------------------------------
static inline Epi epi0() { Epi e{}; e.sign = 1.0f; return e; }

static void gemm(hipStream_t s, const __bf16* A, const __bf16* Bq,
                 int batch, int M, int Nn, int K,
                 long sAb, long sAm, long sAk,
                 long sBb, long sBk, long sBn, const Epi& e)
{
    int tM = (M + 15) / 16, tN = (Nn + 31) / 32;
    long tiles = (long)batch * tM * tN;
    int blocks = (int)((tiles + 3) / 4);
    if (sAk == 1 && sBk == 1)
        wmma_gemm<0, 0><<<blocks, 128, 0, s>>>(A, Bq, batch, M, Nn, K,
                                               sAb, sAm, sAk, sBb, sBk, sBn, tM, tN, e);
    else if (sAk == 1)
        wmma_gemm<0, 1><<<blocks, 128, 0, s>>>(A, Bq, batch, M, Nn, K,
                                               sAb, sAm, sAk, sBb, sBk, sBn, tM, tN, e);
    else
        wmma_gemm<1, 1><<<blocks, 128, 0, s>>>(A, Bq, batch, M, Nn, K,
                                               sAb, sAm, sAk, sBb, sBk, sBn, tM, tN, e);
}

extern "C" void kernel_launch(void* const* d_in, const int* in_sizes, int n_in,
                              void* d_out, int out_size, void* d_ws, size_t ws_size,
                              hipStream_t stream)
{
    (void)in_sizes; (void)n_in; (void)out_size; (void)ws_size;
    const float* xyz  = (const float*)d_in[0];
    const float* clsl = (const float*)d_in[1];
    const float* Amat = (const float*)d_in[2];
    const float* Bmat = (const float*)d_in[3];
    const float* W1r  = (const float*)d_in[4];
    const float* b1r  = (const float*)d_in[5];
    const float* W1i  = (const float*)d_in[6];
    const float* b1i  = (const float*)d_in[7];
    const float* W2r  = (const float*)d_in[8];
    const float* b2r  = (const float*)d_in[9];
    const float* W2i  = (const float*)d_in[10];
    const float* b2i  = (const float*)d_in[11];
    const float* lbl  = (const float*)d_in[12];
    const float* ftw1 = (const float*)d_in[13];
    const float* ftg1 = (const float*)d_in[14];
    const float* ftb1 = (const float*)d_in[15];
    const float* ftw2 = (const float*)d_in[16];
    const float* ftg2 = (const float*)d_in[17];
    const float* ftb2 = (const float*)d_in[18];
    const float* saqk = (const float*)d_in[19];
    const float* savw = (const float*)d_in[20];
    const float* savb = (const float*)d_in[21];
    const float* satw = (const float*)d_in[22];
    const float* satb = (const float*)d_in[23];
    const float* sag  = (const float*)d_in[24];
    const float* sab  = (const float*)d_in[25];
    const float* fusew = (const float*)d_in[26];
    const float* fuseg = (const float*)d_in[27];
    const float* fuseb = (const float*)d_in[28];
    const float* c1w  = (const float*)d_in[29];
    const float* c1b  = (const float*)d_in[30];
    const float* bn1g = (const float*)d_in[31];
    const float* bn1b = (const float*)d_in[32];
    const float* c2w  = (const float*)d_in[33];
    const float* c2b  = (const float*)d_in[34];

    char* base = (char*)d_ws;
    size_t off = 0;
    auto alloc = [&](size_t bytes) -> void* {
        void* p = base + off;
        off = (off + bytes + 255) & ~((size_t)255);
        return p;
    };
    const long ND = (long)NT * DD;      // token-major D-width tensors
    const long NC = (long)NT * CC;

    // bf16 activations
    __bf16* eAr  = (__bf16*)alloc(ND * 2);
    __bf16* eAi  = (__bf16*)alloc(ND * 2);
    __bf16* eBr  = (__bf16*)alloc(ND * 2);
    __bf16* eBi  = (__bf16*)alloc(ND * 2);
    __bf16* inRb = (__bf16*)alloc((size_t)BB * DD * DD * 2);
    __bf16* inIb = (__bf16*)alloc((size_t)BB * DD * DD * 2);
    __bf16* GnR  = (__bf16*)alloc(ND * 2);
    __bf16* GnI  = (__bf16*)alloc(ND * 2);
    __bf16* HrB  = (__bf16*)alloc(ND * 2);
    __bf16* HiB  = (__bf16*)alloc(ND * 2);
    __bf16* xdnB = (__bf16*)alloc(ND * 2);                    // (B,D,N)
    __bf16* xaB  = (__bf16*)alloc(NC * 2);
    __bf16* x0B  = (__bf16*)alloc(NC * 2);
    __bf16* QB   = (__bf16*)alloc((size_t)BB * 32 * NP * 2);
    __bf16* attB = (__bf16*)alloc((size_t)BB * NP * NP * 2);
    __bf16* xvB  = (__bf16*)alloc(NC * 2);
    __bf16* xrB  = (__bf16*)alloc(NC * 2);
    __bf16* xcat = (__bf16*)alloc((size_t)BB * 4 * CC * NP * 2);
    __bf16* xfB  = (__bf16*)alloc(NC * 2);
    __bf16* xc1B = (__bf16*)alloc(NC * 2);
    // bf16 weights
    __bf16* w1rB = (__bf16*)alloc((size_t)DD * DD * 2);
    __bf16* w1iB = (__bf16*)alloc((size_t)DD * DD * 2);
    __bf16* w2rB = (__bf16*)alloc((size_t)DD * DD * 2);
    __bf16* w2iB = (__bf16*)alloc((size_t)DD * DD * 2);
    __bf16* fw1B = (__bf16*)alloc((size_t)CC * DD * 2);
    __bf16* fw2B = (__bf16*)alloc((size_t)CC * CC * 2);
    __bf16* qkB  = (__bf16*)alloc((size_t)4 * 32 * CC * 2);
    __bf16* vwB  = (__bf16*)alloc((size_t)4 * CC * CC * 2);
    __bf16* twB  = (__bf16*)alloc((size_t)4 * CC * CC * 2);
    __bf16* fuB  = (__bf16*)alloc((size_t)CC * 4 * CC * 2);
    __bf16* c1B  = (__bf16*)alloc((size_t)CC * CC * 2);
    __bf16* c2B  = (__bf16*)alloc((size_t)NCLSv * CC * 2);
    // f32 buffers
    float* innR  = (float*)alloc((size_t)BB * DD * DD * 4);
    float* innI  = (float*)alloc((size_t)BB * DD * DD * 4);
    float* tmpF  = (float*)alloc(ND * 4);
    float* Gr    = (float*)alloc(ND * 4);
    float* Gi    = (float*)alloc(ND * 4);
    float* Hr    = (float*)alloc(ND * 4);
    float* Hi    = (float*)alloc(ND * 4);
    float* rw    = (float*)alloc((size_t)BB * DD * 4);
    float* xa    = (float*)alloc(NC * 4);
    float* x0    = (float*)alloc(NC * 4);
    float* energy = (float*)alloc((size_t)BB * NP * NP * 4);
    float* colsum = (float*)alloc((size_t)BB * NP * 4);
    float* xr    = (float*)alloc(NC * 4);
    float* saOut = (float*)alloc((size_t)4 * NC * 4);
    float* c2o   = (float*)alloc((size_t)BB * NCLSv * NP * 4);
    float* nb1i  = (float*)alloc(DD * 4);
    float* nb2i  = (float*)alloc(DD * 4);
    float* ft1S  = (float*)alloc(CC * 4);
    float* ft1B  = (float*)alloc(CC * 4);
    float* ft2S  = (float*)alloc(CC * 4);
    float* ft2B  = (float*)alloc(CC * 4);
    float* saS   = (float*)alloc((size_t)4 * CC * 4);
    float* saFB  = (float*)alloc((size_t)4 * CC * 4);
    float* fuS   = (float*)alloc(CC * 4);
    float* fuFB  = (float*)alloc(CC * 4);
    float* bn1S  = (float*)alloc(CC * 4);
    float* bn1FB = (float*)alloc(CC * 4);

    // ---- prep: phases, weight conversions, fused scale/bias ----------------
    k_phase<<<(int)((ND + 255) / 256), 256, 0, stream>>>(xyz, Amat, Bmat, eAr, eAi, eBr, eBi);
    k_f2bf<<<(DD * DD + 255) / 256, 256, 0, stream>>>(W1r, w1rB, (long)DD * DD);
    k_f2bf<<<(DD * DD + 255) / 256, 256, 0, stream>>>(W1i, w1iB, (long)DD * DD);
    k_f2bf<<<(DD * DD + 255) / 256, 256, 0, stream>>>(W2r, w2rB, (long)DD * DD);
    k_f2bf<<<(DD * DD + 255) / 256, 256, 0, stream>>>(W2i, w2iB, (long)DD * DD);
    k_f2bf<<<(CC * DD + 255) / 256, 256, 0, stream>>>(ftw1, fw1B, (long)CC * DD);
    k_f2bf<<<(CC * CC + 255) / 256, 256, 0, stream>>>(ftw2, fw2B, (long)CC * CC);
    k_f2bf<<<(4 * 32 * CC + 255) / 256, 256, 0, stream>>>(saqk, qkB, (long)4 * 32 * CC);
    k_f2bf<<<(4 * CC * CC + 255) / 256, 256, 0, stream>>>(savw, vwB, (long)4 * CC * CC);
    k_f2bf<<<(4 * CC * CC + 255) / 256, 256, 0, stream>>>(satw, twB, (long)4 * CC * CC);
    k_f2bf<<<(CC * 4 * CC + 255) / 256, 256, 0, stream>>>(fusew, fuB, (long)CC * 4 * CC);
    k_f2bf<<<(CC * CC + 255) / 256, 256, 0, stream>>>(c1w, c1B, (long)CC * CC);
    k_f2bf<<<(NCLSv * CC + 255) / 256, 256, 0, stream>>>(c2w, c2B, (long)NCLSv * CC);
    k_neg<<<1, DD, 0, stream>>>(b1i, nb1i, DD);
    k_neg<<<1, DD, 0, stream>>>(b2i, nb2i, DD);
    k_prep_sb<<<1, CC, 0, stream>>>(ftg1, ftb1, nullptr, ft1S, ft1B, CC);
    k_prep_sb<<<1, CC, 0, stream>>>(ftg2, ftb2, nullptr, ft2S, ft2B, CC);
    k_prep_sb<<<2, 256, 0, stream>>>(sag, sab, satb, saS, saFB, 4 * CC);
    k_prep_sb<<<1, CC, 0, stream>>>(fuseg, fuseb, nullptr, fuS, fuFB, CC);
    k_prep_sb<<<1, CC, 0, stream>>>(bn1g, bn1b, c1b, bn1S, bn1FB, CC);
    k_reweight<<<(BB * DD + 255) / 256, 256, 0, stream>>>(clsl, lbl, rw);

    // ---- VecKM: inner = eB^H eA  (per batch, D x D, K = N) -----------------
    {   // innerR = eBr^T eAr + eBi^T eAi
        Epi e = epi0(); e.C = innR; e.cB = (long)DD * DD;
        gemm(stream, eBr, eAr, BB, DD, DD, NP, (long)NP*DD, 1, DD, (long)NP*DD, DD, 1, e);
        e.prevPre = innR; e.ppB = (long)DD * DD; e.Cbf = inRb; e.cbfB = (long)DD * DD;
        gemm(stream, eBi, eAi, BB, DD, DD, NP, (long)NP*DD, 1, DD, (long)NP*DD, DD, 1, e);
    }
    {   // innerI = eBr^T eAi - eBi^T eAr
        Epi e = epi0(); e.C = tmpF; e.cB = (long)DD * DD;
        gemm(stream, eBr, eAi, BB, DD, DD, NP, (long)NP*DD, 1, DD, (long)NP*DD, DD, 1, e);
        Epi e2 = epi0(); e2.sign = -1.0f; e2.prevPre = tmpF; e2.ppB = (long)DD * DD;
        e2.C = innI; e2.cB = (long)DD * DD; e2.Cbf = inIb; e2.cbfB = (long)DD * DD;
        gemm(stream, eBi, eAr, BB, DD, DD, NP, (long)NP*DD, 1, DD, (long)NP*DD, DD, 1, e2);
    }
    // ---- G = eB @ inner (complex, per batch, N x D, K = D) -----------------
    {   // Gr = eBr*innR - eBi*innI
        Epi e = epi0(); e.C = Gr; e.cB = (long)NP * DD;
        gemm(stream, eBr, inRb, BB, NP, DD, DD, (long)NP*DD, DD, 1, (long)DD*DD, DD, 1, e);
        Epi e2 = epi0(); e2.sign = -1.0f; e2.prevPre = Gr; e2.ppB = (long)NP * DD;
        e2.C = Gr; e2.cB = (long)NP * DD;
        gemm(stream, eBi, inIb, BB, NP, DD, DD, (long)NP*DD, DD, 1, (long)DD*DD, DD, 1, e2);
        // Gi = eBr*innI + eBi*innR
        Epi e3 = epi0(); e3.C = Gi; e3.cB = (long)NP * DD;
        gemm(stream, eBr, inIb, BB, NP, DD, DD, (long)NP*DD, DD, 1, (long)DD*DD, DD, 1, e3);
        Epi e4 = epi0(); e4.prevPre = Gi; e4.ppB = (long)NP * DD;
        e4.C = Gi; e4.cB = (long)NP * DD;
        gemm(stream, eBi, inRb, BB, NP, DD, DD, (long)NP*DD, DD, 1, (long)DD*DD, DD, 1, e4);
    }
    k_vnorm<<<NT, 256, 0, stream>>>(Gr, Gi, GnR, GnI);

    // ---- ComplexLinear 1 + ComplexReLU (token-major, batch folded into M) --
    {   // re = GnR*W1r^T + b1r - (GnI*W1i^T + b1i), relu
        Epi e = epi0(); e.bias = b1r; e.C = Hr; e.cB = ND;
        gemm(stream, GnR, w1rB, 1, NT, DD, DD, 0, DD, 1, 0, 1, DD, e);
        Epi e2 = epi0(); e2.sign = -1.0f; e2.prevPre = Hr; e2.ppB = ND;
        e2.bias = nb1i; e2.relu = 1; e2.C = Hr; e2.cB = ND; e2.Cbf = HrB; e2.cbfB = ND;
        gemm(stream, GnI, w1iB, 1, NT, DD, DD, 0, DD, 1, 0, 1, DD, e2);
        // im = GnI*W1r^T + b1r + GnR*W1i^T + b1i, relu
        Epi e3 = epi0(); e3.bias = b1r; e3.C = tmpF; e3.cB = ND;
        gemm(stream, GnI, w1rB, 1, NT, DD, DD, 0, DD, 1, 0, 1, DD, e3);
        Epi e4 = epi0(); e4.prevPre = tmpF; e4.ppB = ND;
        e4.bias = b1i; e4.relu = 1; e4.C = Hi; e4.cB = ND; e4.Cbf = HiB; e4.cbfB = ND;
        gemm(stream, GnR, w1iB, 1, NT, DD, DD, 0, DD, 1, 0, 1, DD, e4);
    }
    // ---- ComplexLinear 2 (output reuses Gr/Gi as Fr/Fi) --------------------
    {
        Epi e = epi0(); e.bias = b2r; e.C = Gr; e.cB = ND;
        gemm(stream, HrB, w2rB, 1, NT, DD, DD, 0, DD, 1, 0, 1, DD, e);
        Epi e2 = epi0(); e2.sign = -1.0f; e2.prevPre = Gr; e2.ppB = ND;
        e2.bias = nb2i; e2.C = Gr; e2.cB = ND;
        gemm(stream, HiB, w2iB, 1, NT, DD, DD, 0, DD, 1, 0, 1, DD, e2);
        Epi e3 = epi0(); e3.bias = b2r; e3.C = tmpF; e3.cB = ND;
        gemm(stream, HiB, w2rB, 1, NT, DD, DD, 0, DD, 1, 0, 1, DD, e3);
        Epi e4 = epi0(); e4.prevPre = tmpF; e4.ppB = ND;
        e4.bias = b2i; e4.C = Gi; e4.cB = ND;
        gemm(stream, HrB, w2iB, 1, NT, DD, DD, 0, DD, 1, 0, 1, DD, e4);
    }
    k_magrw<<<(int)((ND + 255) / 256), 256, 0, stream>>>(Gr, Gi, rw, xdnB);

    // ---- feature transform convs (channel-major, per batch) ----------------
    {
        Epi e = epi0(); e.scale = ft1S; e.bias = ft1B; e.chRow = 1; e.relu = 1;
        e.C = xa; e.cB = (long)CC * NP; e.Cbf = xaB; e.cbfB = (long)CC * NP;
        gemm(stream, fw1B, xdnB, BB, CC, NP, DD, 0, DD, 1, (long)DD*NP, NP, 1, e);
        Epi e2 = epi0(); e2.scale = ft2S; e2.bias = ft2B; e2.chRow = 1;
        e2.C = x0; e2.cB = (long)CC * NP; e2.Cbf = x0B; e2.cbfB = (long)CC * NP;
        gemm(stream, fw2B, xaB, BB, CC, NP, CC, 0, CC, 1, (long)CC*NP, NP, 1, e2);
    }

    // ---- 4 offset-attention layers ----------------------------------------
    for (int i = 0; i < 4; ++i) {
        const float*  pF   = (i == 0) ? x0 : saOut + (size_t)(i - 1) * NC;
        const __bf16* pB   = (i == 0) ? x0B : xcat + (size_t)(i - 1) * CC * NP;
        const long    pBst = (i == 0) ? (long)CC * NP : (long)4 * CC * NP;
        // Q (32 x N) = qkW @ x
        Epi eq = epi0(); eq.Cbf = QB; eq.cbfB = (long)32 * NP;
        gemm(stream, qkB + (size_t)i * 32 * CC, pB, BB, 32, NP, CC,
             0, CC, 1, pBst, NP, 1, eq);
        // energy (N x N) = Q^T Q
        Epi ee = epi0(); ee.C = energy; ee.cB = (long)NP * NP;
        gemm(stream, QB, QB, BB, NP, NP, 32,
             (long)32 * NP, 1, NP, (long)32 * NP, NP, 1, ee);
        k_softmax_rows<<<BB * NP, 256, 0, stream>>>(energy);
        k_colsum<<<dim3(NP / 256, BB), 256, 0, stream>>>(energy, colsum);
        k_renorm<<<(int)(((long)BB * NP * NP + 255) / 256), 256, 0, stream>>>(energy, colsum, attB);
        // xv (C x N) = vW @ x + vb
        Epi ev = epi0(); ev.bias = savb + (size_t)i * CC; ev.chRow = 1;
        ev.Cbf = xvB; ev.cbfB = (long)CC * NP;
        gemm(stream, vwB + (size_t)i * CC * CC, pB, BB, CC, NP, CC,
             0, CC, 1, pBst, NP, 1, ev);
        // xr = x - xv @ attn
        Epi er = epi0(); er.sign = -1.0f; er.prevPre = pF; er.ppB = (long)CC * NP;
        er.C = xr; er.cB = (long)CC * NP; er.Cbf = xrB; er.cbfB = (long)CC * NP;
        gemm(stream, xvB, attB, BB, CC, NP, NP,
             (long)CC * NP, NP, 1, (long)NP * NP, NP, 1, er);
        // out = x + relu(bn(tW @ xr + tb))  -> saOut[i] f32, xcat slice bf16
        Epi et = epi0(); et.scale = saS + (size_t)i * CC; et.bias = saFB + (size_t)i * CC;
        et.chRow = 1; et.relu = 1; et.prevPost = pF; et.poB = (long)CC * NP;
        et.C = saOut + (size_t)i * NC; et.cB = (long)CC * NP;
        et.Cbf = xcat + (size_t)i * CC * NP; et.cbfB = (long)4 * CC * NP;
        gemm(stream, twB + (size_t)i * CC * CC, xrB, BB, CC, NP, CC,
             0, CC, 1, (long)CC * NP, NP, 1, et);
    }

    // ---- fuse -> conv1 -> conv2 -> log_softmax -----------------------------
    {
        Epi ef = epi0(); ef.scale = fuS; ef.bias = fuFB; ef.chRow = 1; ef.relu = 1;
        ef.Cbf = xfB; ef.cbfB = (long)CC * NP;
        gemm(stream, fuB, xcat, BB, CC, NP, 4 * CC,
             0, (long)4 * CC, 1, (long)4 * CC * NP, NP, 1, ef);
        Epi e1 = epi0(); e1.scale = bn1S; e1.bias = bn1FB; e1.chRow = 1; e1.relu = 1;
        e1.Cbf = xc1B; e1.cbfB = (long)CC * NP;
        gemm(stream, c1B, xfB, BB, CC, NP, CC, 0, CC, 1, (long)CC * NP, NP, 1, e1);
        Epi e2 = epi0(); e2.bias = c2b; e2.chRow = 1;
        e2.C = c2o; e2.cB = (long)NCLSv * NP;
        gemm(stream, c2B, xc1B, BB, NCLSv, NP, CC, 0, CC, 1, (long)CC * NP, NP, 1, e2);
    }
    k_logsm<<<(NT + 255) / 256, 256, 0, stream>>>(c2o, (float*)d_out);
}
